// SpaMamba_75316546503286
// MI455X (gfx1250) — compile-verified, matrix operand order checked by reading, not run on hardware
//
#include <hip/hip_runtime.h>
#include <hip/hip_bf16.h>
#include <math.h>

// ---- problem constants ----
#define LSEQ   8192
#define NBATCH 2
#define NTOK   (NBATCH * LSEQ)   // 16384
#define DMODEL 128
#define DINNER 256
#define DSTATE 16
#define DTRANK 8
#define NCHUNK 32                 // chunks per batch
#define LCHUNK (LSEQ / NCHUNK)    // 256
#define NTASK  (NBATCH * DINNER * NCHUNK) // 16384 chunk-tasks

typedef __attribute__((ext_vector_type(16))) __bf16 v16bf;
typedef __attribute__((ext_vector_type(8)))  float  v8f;

__device__ __forceinline__ float sigmoidf_(float v) {
    return 1.0f / (1.0f + __expf(-v));
}

// split f32 into bf16 hi + bf16 lo (residual), ~16 mantissa bits total
__device__ __forceinline__ void split_hl(__bf16* hd, __bf16* ld, float v) {
    __bf16 h = (__bf16)v;
    *hd = h;
    *ld = (__bf16)(v - (float)h);
}

// build one 16-bit WMMA fragment from an LDS row (A or B, symmetric layout)
__device__ __forceinline__ v16bf frag16(const __bf16* __restrict__ p, int half) {
    v16bf f;
#pragma unroll
    for (int i = 0; i < 16; ++i) {
        int k = ((i >> 3) << 4) + half * 8 + (i & 7);
        f[i] = p[k];
    }
    return f;
}

// ---------------------------------------------------------------------------
// K1: in-projection GEMM  xz[t][e] = sum_k x[t][k] * W_in[e][k]
//     split-bf16 WMMA 16x16x32 (hi*hi + hi*lo + lo*hi), f32 accumulate.
//     Tile: 32 tokens x 64 channels per workgroup; 8 waves = 2x4 subtiles.
// ---------------------------------------------------------------------------
__global__ __launch_bounds__(256) void k_inproj(const float* __restrict__ x,
                                                const float* __restrict__ W_in,
                                                float* __restrict__ xi,
                                                float* __restrict__ zb) {
    const int m0 = blockIdx.x * 32;   // token tile base
    const int n0 = blockIdx.y * 64;   // output channel tile base (0..511)
    __shared__ __bf16 xsh[32 * 136];
    __shared__ __bf16 xsl[32 * 136];
    __shared__ __bf16 wsh[64 * 136];
    __shared__ __bf16 wsl[64 * 136];
    const int tid = threadIdx.x;

    // x tile: 32 rows x 128 k, float4 per lane
    for (int idx = tid; idx < 32 * 32; idx += 256) {
        int m = idx >> 5, kv = (idx & 31) << 2;
        const float4 v = *(const float4*)(x + (m0 + m) * DMODEL + kv);
        int o = m * 136 + kv;
        split_hl(&xsh[o + 0], &xsl[o + 0], v.x);
        split_hl(&xsh[o + 1], &xsl[o + 1], v.y);
        split_hl(&xsh[o + 2], &xsl[o + 2], v.z);
        split_hl(&xsh[o + 3], &xsl[o + 3], v.w);
    }
    // W tile: 64 rows x 128 k
    for (int idx = tid; idx < 64 * 32; idx += 256) {
        int n = idx >> 5, kv = (idx & 31) << 2;
        const float4 v = *(const float4*)(W_in + (n0 + n) * DMODEL + kv);
        int o = n * 136 + kv;
        split_hl(&wsh[o + 0], &wsl[o + 0], v.x);
        split_hl(&wsh[o + 1], &wsl[o + 1], v.y);
        split_hl(&wsh[o + 2], &wsl[o + 2], v.z);
        split_hl(&wsh[o + 3], &wsl[o + 3], v.w);
    }
    __syncthreads();

    const int wave = tid >> 5, lane = tid & 31;
    const int half = lane >> 4, col = lane & 15;
    const int msub = (wave >> 2) * 16;   // 2 M sub-tiles
    const int nsub = (wave & 3) * 16;    // 4 N sub-tiles

    v8f acc = {};
#pragma unroll
    for (int kk = 0; kk < 128; kk += 32) {
        v16bf ah = frag16(&xsh[(msub + col) * 136 + kk], half);
        v16bf al = frag16(&xsl[(msub + col) * 136 + kk], half);
        v16bf bh = frag16(&wsh[(nsub + col) * 136 + kk], half);
        v16bf bl = frag16(&wsl[(nsub + col) * 136 + kk], half);
        acc = __builtin_amdgcn_wmma_f32_16x16x32_bf16(false, ah, false, bh, (short)0, acc, false, false);
        acc = __builtin_amdgcn_wmma_f32_16x16x32_bf16(false, ah, false, bl, (short)0, acc, false, false);
        acc = __builtin_amdgcn_wmma_f32_16x16x32_bf16(false, al, false, bh, (short)0, acc, false, false);
    }

    const int ch = n0 + nsub + col;                 // 0..511
    float* dst = (ch < DINNER) ? (xi + ch) : (zb + (ch - DINNER));
#pragma unroll
    for (int r = 0; r < 8; ++r) {
        int row = m0 + msub + half * 8 + r;
        dst[row * DINNER] = acc[r];
    }
}

// ---------------------------------------------------------------------------
// K2: causal depthwise conv (width 4) + SiLU
// ---------------------------------------------------------------------------
__global__ __launch_bounds__(256) void k_conv(const float* __restrict__ xi,
                                              const float* __restrict__ cw,
                                              const float* __restrict__ cb,
                                              float* __restrict__ xc) {
    int idx = blockIdx.x * 256 + threadIdx.x;      // over NTOK*DINNER
    int d = idx & (DINNER - 1);
    int t = idx >> 8;
    int l = t & (LSEQ - 1);
    float acc = cb[d];
#pragma unroll
    for (int j = 0; j < 4; ++j) {
        int ls = l - 3 + j;
        if (ls >= 0) acc = fmaf(cw[d * 4 + j], xi[(t - 3 + j) * DINNER + d], acc);
    }
    xc[idx] = acc * sigmoidf_(acc);
}

// ---------------------------------------------------------------------------
// K3: x-projection  xdbl[t][e] = sum_k xc[t][k] * W_xproj[e][k]   (e < 40)
//     split outputs into dt_r / Bm / Cm buffers.
// ---------------------------------------------------------------------------
__global__ __launch_bounds__(256) void k_xproj(const float* __restrict__ xc,
                                               const float* __restrict__ Wx,
                                               float* __restrict__ dtr,
                                               float* __restrict__ Bm,
                                               float* __restrict__ Cm) {
    int idx = blockIdx.x * 256 + threadIdx.x;
    if (idx >= NTOK * 40) return;
    int e = idx % 40;
    int t = idx / 40;
    const float* row = xc + t * DINNER;
    const float* w   = Wx + e * DINNER;
    float acc = 0.f;
#pragma unroll 8
    for (int k = 0; k < DINNER; ++k) acc = fmaf(row[k], w[k], acc);
    if (e < DTRANK)            dtr[t * DTRANK + e] = acc;
    else if (e < DTRANK + 16)  Bm[t * DSTATE + (e - DTRANK)] = acc;
    else                       Cm[t * DSTATE + (e - DTRANK - DSTATE)] = acc;
}

// ---------------------------------------------------------------------------
// K4: dt = softplus(dt_r @ W_dt^T + b_dt)
// ---------------------------------------------------------------------------
__global__ __launch_bounds__(256) void k_dt(const float* __restrict__ dtr,
                                            const float* __restrict__ Wdt,
                                            const float* __restrict__ bdt,
                                            float* __restrict__ dt) {
    int idx = blockIdx.x * 256 + threadIdx.x;      // NTOK*DINNER
    int d = idx & (DINNER - 1);
    int t = idx >> 8;
    float acc = bdt[d];
#pragma unroll
    for (int r = 0; r < DTRANK; ++r) acc = fmaf(dtr[t * DTRANK + r], Wdt[d * DTRANK + r], acc);
    dt[idx] = (acc > 20.f) ? acc : log1pf(__expf(acc));
}

// ---------------------------------------------------------------------------
// K5: scan phase 1 — per-chunk local scan (zero init): chunk product + state.
//     lane = state index n (16-lane groups); 2 chunk-tasks per wave.
// ---------------------------------------------------------------------------
__global__ __launch_bounds__(256) void k_scan1(const float* __restrict__ dt,
                                               const float* __restrict__ xc,
                                               const float* __restrict__ Bm,
                                               const float* __restrict__ A_log,
                                               float* __restrict__ Pa,
                                               float* __restrict__ Hloc) {
    int tid = blockIdx.x * 256 + threadIdx.x;
    int n = tid & (DSTATE - 1);
    int T = tid >> 4;                    // chunk-task 0..NTASK-1
    int c = T & (NCHUNK - 1);
    int G = T >> 5;                      // channel 0..511
    int d = G & (DINNER - 1), b = G >> 8;
    float A = -__expf(A_log[d * DSTATE + n]);
    float h = 0.f, p = 1.f;
    int tok = b * LSEQ + c * LCHUNK;
    for (int s = 0; s < LCHUNK; ++s, ++tok) {
        float dtv = dt[tok * DINNER + d];
        float a   = __expf(dtv * A);
        float xv  = dtv * Bm[tok * DSTATE + n] * xc[tok * DINNER + d];
        h = fmaf(a, h, xv);
        p *= a;
    }
    Pa[T * DSTATE + n]   = p;
    Hloc[T * DSTATE + n] = h;
}

// ---------------------------------------------------------------------------
// K6: scan phase 2 — tiny sequential scan over chunk boundaries.
// ---------------------------------------------------------------------------
__global__ __launch_bounds__(256) void k_scan2(const float* __restrict__ Pa,
                                               const float* __restrict__ Hloc,
                                               float* __restrict__ Hin) {
    int tid = blockIdx.x * 256 + threadIdx.x;       // 512*16 lanes
    if (tid >= NBATCH * DINNER * DSTATE) return;
    int n = tid & (DSTATE - 1), G = tid >> 4;
    float h = 0.f;
    for (int c = 0; c < NCHUNK; ++c) {
        int T = G * NCHUNK + c;
        Hin[T * DSTATE + n] = h;
        h = fmaf(Pa[T * DSTATE + n], h, Hloc[T * DSTATE + n]);
    }
}

// ---------------------------------------------------------------------------
// K7: scan phase 3 — re-scan each chunk with correct entry state, emit y.
//     Cross-lane sum over the 16 state lanes via __shfl_xor (stays in group).
// ---------------------------------------------------------------------------
__global__ __launch_bounds__(256) void k_scan3(const float* __restrict__ dt,
                                               const float* __restrict__ xc,
                                               const float* __restrict__ Bm,
                                               const float* __restrict__ Cm,
                                               const float* __restrict__ A_log,
                                               const float* __restrict__ Hin,
                                               float* __restrict__ ys) {
    int tid = blockIdx.x * 256 + threadIdx.x;
    int n = tid & (DSTATE - 1);
    int T = tid >> 4;
    int c = T & (NCHUNK - 1);
    int G = T >> 5;
    int d = G & (DINNER - 1), b = G >> 8;
    float A = -__expf(A_log[d * DSTATE + n]);
    float h = Hin[T * DSTATE + n];
    int tok = b * LSEQ + c * LCHUNK;
    for (int s = 0; s < LCHUNK; ++s, ++tok) {
        float dtv = dt[tok * DINNER + d];
        float a   = __expf(dtv * A);
        float xv  = dtv * Bm[tok * DSTATE + n] * xc[tok * DINNER + d];
        h = fmaf(a, h, xv);
        float yp = h * Cm[tok * DSTATE + n];
        yp += __shfl_xor(yp, 1);
        yp += __shfl_xor(yp, 2);
        yp += __shfl_xor(yp, 4);
        yp += __shfl_xor(yp, 8);
        if (n == 0) ys[tok * DINNER + d] = yp;
    }
}

// ---------------------------------------------------------------------------
// K8: out-projection GEMM with fused gate/skip epilogue + residual.
//     y = (ys + D*xc) * silu(z);  out = y @ W_out^T + x   (split-bf16 WMMA)
// ---------------------------------------------------------------------------
__global__ __launch_bounds__(256) void k_outproj(const float* __restrict__ ys,
                                                 const float* __restrict__ xc,
                                                 const float* __restrict__ zb,
                                                 const float* __restrict__ Dv,
                                                 const float* __restrict__ W_out,
                                                 const float* __restrict__ x,
                                                 float* __restrict__ out) {
    const int m0 = blockIdx.x * 32;
    const int n0 = blockIdx.y * 64;     // 0 or 64 (DMODEL=128)
    __shared__ __bf16 ysh[32 * 264];
    __shared__ __bf16 ysl[32 * 264];
    __shared__ __bf16 wshh[64 * 264];
    __shared__ __bf16 wshl[64 * 264];
    const int tid = threadIdx.x;

    // fused gate/skip epilogue into the A tile (32 tokens x 256 d), float4 lanes
    for (int idx = tid; idx < 32 * 64; idx += 256) {
        int m = idx >> 6, d4 = (idx & 63) << 2;
        int t = m0 + m;
        const float4 yv4 = *(const float4*)(ys + t * DINNER + d4);
        const float4 xc4 = *(const float4*)(xc + t * DINNER + d4);
        const float4 zv4 = *(const float4*)(zb + t * DINNER + d4);
        const float4 dv4 = *(const float4*)(Dv + d4);
        float yv[4];
        yv[0] = (yv4.x + dv4.x * xc4.x) * (zv4.x * sigmoidf_(zv4.x));
        yv[1] = (yv4.y + dv4.y * xc4.y) * (zv4.y * sigmoidf_(zv4.y));
        yv[2] = (yv4.z + dv4.z * xc4.z) * (zv4.z * sigmoidf_(zv4.z));
        yv[3] = (yv4.w + dv4.w * xc4.w) * (zv4.w * sigmoidf_(zv4.w));
        int o = m * 264 + d4;
#pragma unroll
        for (int j = 0; j < 4; ++j) split_hl(&ysh[o + j], &ysl[o + j], yv[j]);
    }
    for (int idx = tid; idx < 64 * 64; idx += 256) {
        int n = idx >> 6, d4 = (idx & 63) << 2;
        const float4 v = *(const float4*)(W_out + (n0 + n) * DINNER + d4);
        int o = n * 264 + d4;
        split_hl(&wshh[o + 0], &wshl[o + 0], v.x);
        split_hl(&wshh[o + 1], &wshl[o + 1], v.y);
        split_hl(&wshh[o + 2], &wshl[o + 2], v.z);
        split_hl(&wshh[o + 3], &wshl[o + 3], v.w);
    }
    __syncthreads();

    const int wave = tid >> 5, lane = tid & 31;
    const int half = lane >> 4, col = lane & 15;
    const int msub = (wave >> 2) * 16;
    const int nsub = (wave & 3) * 16;

    v8f acc = {};
#pragma unroll
    for (int kk = 0; kk < 256; kk += 32) {
        v16bf ah = frag16(&ysh[(msub + col) * 264 + kk], half);
        v16bf al = frag16(&ysl[(msub + col) * 264 + kk], half);
        v16bf bh = frag16(&wshh[(nsub + col) * 264 + kk], half);
        v16bf bl = frag16(&wshl[(nsub + col) * 264 + kk], half);
        acc = __builtin_amdgcn_wmma_f32_16x16x32_bf16(false, ah, false, bh, (short)0, acc, false, false);
        acc = __builtin_amdgcn_wmma_f32_16x16x32_bf16(false, ah, false, bl, (short)0, acc, false, false);
        acc = __builtin_amdgcn_wmma_f32_16x16x32_bf16(false, al, false, bh, (short)0, acc, false, false);
    }

    const int cch = n0 + nsub + col;   // output model channel
#pragma unroll
    for (int r = 0; r < 8; ++r) {
        int row = m0 + msub + half * 8 + r;
        out[row * DMODEL + cch] = acc[r] + x[row * DMODEL + cch];
    }
}

// ---------------------------------------------------------------------------
extern "C" void kernel_launch(void* const* d_in, const int* in_sizes, int n_in,
                              void* d_out, int out_size, void* d_ws, size_t ws_size,
                              hipStream_t stream) {
    const float* x      = (const float*)d_in[0];
    const float* W_in   = (const float*)d_in[1];
    const float* conv_w = (const float*)d_in[2];
    const float* conv_b = (const float*)d_in[3];
    const float* W_xprj = (const float*)d_in[4];
    const float* W_dt   = (const float*)d_in[5];
    const float* b_dt   = (const float*)d_in[6];
    const float* A_log  = (const float*)d_in[7];
    const float* Dv     = (const float*)d_in[8];
    const float* W_out  = (const float*)d_in[9];
    float* out = (float*)d_out;

    char* w = (char*)d_ws;
    size_t off = 0;
    auto carve = [&](size_t bytes) -> float* {
        float* p = (float*)(w + off);
        off = (off + bytes + 255) & ~((size_t)255);
        return p;
    };
    float* xi   = carve((size_t)NTOK * DINNER * 4);   // 16 MB
    float* zb   = carve((size_t)NTOK * DINNER * 4);   // 16 MB
    float* xc   = carve((size_t)NTOK * DINNER * 4);   // 16 MB
    float* dt   = carve((size_t)NTOK * DINNER * 4);   // 16 MB
    float* dtr  = carve((size_t)NTOK * DTRANK * 4);   // 0.5 MB
    float* Bm   = carve((size_t)NTOK * DSTATE * 4);   // 1 MB
    float* Cm   = carve((size_t)NTOK * DSTATE * 4);   // 1 MB
    float* Pa   = carve((size_t)NTASK * DSTATE * 4);  // 1 MB
    float* Hloc = carve((size_t)NTASK * DSTATE * 4);  // 1 MB
    float* Hin  = carve((size_t)NTASK * DSTATE * 4);  // 1 MB
    float* ysb  = xi;   // xi is dead after the conv; reuse for scan output

    // 1) in-projection (split-bf16 WMMA)
    k_inproj<<<dim3(NTOK / 32, 512 / 64), 256, 0, stream>>>(x, W_in, xi, zb);
    // 2) depthwise causal conv + SiLU
    k_conv<<<(NTOK * DINNER) / 256, 256, 0, stream>>>(xi, conv_w, conv_b, xc);
    // 3) x-projection -> dt_r, B, C
    k_xproj<<<(NTOK * 40 + 255) / 256, 256, 0, stream>>>(xc, W_xprj, dtr, Bm, Cm);
    // 4) dt = softplus(low-rank proj + bias)
    k_dt<<<(NTOK * DINNER) / 256, 256, 0, stream>>>(dtr, W_dt, b_dt, dt);
    // 5-7) chunked selective scan
    k_scan1<<<(NTASK * DSTATE) / 256, 256, 0, stream>>>(dt, xc, Bm, A_log, Pa, Hloc);
    k_scan2<<<(NBATCH * DINNER * DSTATE + 255) / 256, 256, 0, stream>>>(Pa, Hloc, Hin);
    k_scan3<<<(NTASK * DSTATE) / 256, 256, 0, stream>>>(dt, xc, Bm, Cm, A_log, Hin, ysb);
    // 8) gate + skip + out-projection (split-bf16 WMMA) + residual
    k_outproj<<<dim3(NTOK / 32, DMODEL / 64), 256, 0, stream>>>(ysb, xc, zb, Dv, W_out, x, out);
    (void)in_sizes; (void)n_in; (void)out_size; (void)ws_size;
}
